// DyGCGRUCell_57114475102830
// MI455X (gfx1250) — compile-verified, matrix operand order checked by reading, not run on hardware
//
#include <hip/hip_runtime.h>
#include <hip/hip_bf16.h>

// Problem dims
constexpr int NN   = 8192;
constexpr int INW  = 64;
constexpr int HID  = 64;
constexpr int NH   = 4;
constexpr int TOT  = 128;
constexpr int DH   = 32;

typedef __attribute__((ext_vector_type(16))) __bf16 v16bf;
typedef __attribute__((ext_vector_type(8)))  float  v8f;
typedef __attribute__((ext_vector_type(8)))  __bf16 bf16x8;

// Load a 16x32 bf16 tile (row stride in elements) into the WMMA A/B operand layout:
// lane l (m = l&15, hi = l>>4) holds row/col m, K in {hi*8..+7} U {hi*8+16..+23}.
__device__ inline v16bf load_tile(const __bf16* base, int stride, int lane) {
  int m  = lane & 15;
  int hi = lane >> 4;
  const __bf16* p = base + (size_t)m * stride + hi * 8;
  bf16x8 lo = *(const bf16x8*)(p);
  bf16x8 hv = *(const bf16x8*)(p + 16);
  v16bf a;
#pragma unroll
  for (int i = 0; i < 8; ++i) { a[i] = lo[i]; a[i + 8] = hv[i]; }
  return a;
}

// ---------------- Kernel 1a: xt = concat(input, hx)^T -> bf16 [TOT][NN] ----------------
__global__ void concat_t_kernel(const float* __restrict__ in, const float* __restrict__ hx,
                                __bf16* __restrict__ xt) {
  int n = blockIdx.x;
  int t = threadIdx.x; // 128
  float v = (t < INW) ? in[(size_t)n * INW + t] : hx[(size_t)n * HID + (t - INW)];
  xt[(size_t)t * NN + n] = (__bf16)v;
}

// ---------------- Kernel 1b: q,k projections; q pre-scaled by log2(e)/sqrt(DH) ----------------
// Softmax is base-invariant, so scoring in the log2 domain lets the attention kernel use
// v_exp_f32 (exp2) directly with no per-element multiply.
__global__ void qk_kernel(const float* __restrict__ in, const float* __restrict__ hx,
                          const float* __restrict__ Wq, const float* __restrict__ Wk,
                          __bf16* __restrict__ qb, __bf16* __restrict__ kb) {
  int n = blockIdx.x;
  int t = threadIdx.x;       // 128 threads: h = t>>5, d = t&31
  int h = t >> 5, d = t & 31;
  __shared__ float sx[TOT];
  sx[t] = (t < INW) ? in[(size_t)n * INW + t] : hx[(size_t)n * HID + (t - INW)];
  __syncthreads();
  float q = 0.f, k = 0.f;
#pragma unroll 4
  for (int j = 0; j < TOT; ++j) {
    float xv = sx[j];
    q = fmaf(xv, Wq[((size_t)h * TOT + j) * DH + d], q);
    k = fmaf(xv, Wk[((size_t)h * TOT + j) * DH + d], k);
  }
  q *= 0.25502767704977545f; // (1/sqrt(32)) * log2(e)
  qb[((size_t)h * NN + n) * DH + d] = (__bf16)q;
  kb[((size_t)h * NN + n) * DH + d] = (__bf16)k;
}

// ---------------- Kernel 2: fused attention, out = mean_h softmax(q k^T) @ V ----------------
// Transposed formulation: S^T = k_tile x q^T. The C-layout of the two stacked 16x16 S^T
// tiles IS the 32x16 B-operand layout, so exp2(S) feeds the PV WMMAs directly from
// registers: O^T[dim][query] += V^T_chunk(16x32) x P~(32x16). No LDS in the inner loop.
// 8 waves = 2 query tiles x 4 heads per block to halve L2 traffic for k^T / V^T streams.
__global__ __launch_bounds__(256) void flash_kernel(const __bf16* __restrict__ qg,
                                                    const __bf16* __restrict__ kg,
                                                    const __bf16* __restrict__ vt,
                                                    float* __restrict__ og) {
  __shared__ __align__(16) float sO[2][16][TOT];  // per-query-half head accumulation (16 KB)

  const int tid  = threadIdx.x;
  const int w    = tid >> 5;
  const int head = w & 3;
  const int qh   = w >> 2;       // which 16-query half of the block's 32 queries
  const int lane = tid & 31;
  const int qbase = blockIdx.x * 32 + qh * 16;

  // B operand: q tile for this head (resident all loop long); column n = q row n
  v16bf bq = load_tile(qg + ((size_t)head * NN + qbase) * DH, DH, lane);

  v8f OT[8];                    // O^T tiles: dims c*16.. x queries
#pragma unroll
  for (int c = 0; c < 8; ++c) { v8f z = {}; OT[c] = z; }
  float lpart = 0.f;            // per-lane partial row-sum (16 of 32 keys of query lane&15)

  const __bf16* kbase = kg + (size_t)head * NN * DH;

  for (int kt = 0; kt < NN; kt += 32) {
    __builtin_prefetch(kbase + (size_t)(kt + 32) * DH + lane * 32, 0, 3);

    // S^T tiles: T0 = keys kt..kt+15, T1 = keys kt+16..kt+31 (16 keys x 16 queries each)
    v16bf a0 = load_tile(kbase + (size_t)kt * DH, DH, lane);
    v16bf a1 = load_tile(kbase + (size_t)(kt + 16) * DH, DH, lane);
    v8f zero = {};
    v8f T0 = __builtin_amdgcn_wmma_f32_16x16x32_bf16(false, a0, false, bq, (short)0, zero, false, false);
    v8f T1 = __builtin_amdgcn_wmma_f32_16x16x32_bf16(false, a1, false, bq, (short)0, zero, false, false);

    // P~ = exp2(S^T) packed to bf16 in-register: already the 32x16 B-operand layout
    v16bf pb;
    float tsum = 0.f;
#pragma unroll
    for (int e = 0; e < 8; ++e) {
      float p0 = __builtin_amdgcn_exp2f(T0[e]);
      float p1 = __builtin_amdgcn_exp2f(T1[e]);
      tsum += p0 + p1;
      pb[e]     = (__bf16)p0;
      pb[e + 8] = (__bf16)p1;
    }
    lpart += tsum;

    // O^T(128x16) += V^T(128x32-chunk) x P~(32x16); V^T streamed from L2-resident global
#pragma unroll
    for (int c = 0; c < 8; ++c) {
      v16bf av = load_tile(vt + (size_t)(c * 16) * NN + kt, NN, lane);
      OT[c] = __builtin_amdgcn_wmma_f32_16x16x32_bf16(false, av, false, pb, (short)0, OT[c], false, false);
    }
  }

  // Row sum: lane n and lane n+16 hold complementary key halves of query n
  lpart += __shfl_xor(lpart, 16, 32);
  float inv = 1.f / lpart;

  // Accumulate heads into sO[qh] in 4 deterministic barrier-separated phases.
  // O^T C-layout: lane holds query (lane&15), dims c*16 + 8*(lane>>4) + e (contiguous).
  const int qn  = lane & 15;
  const int dhi = (lane >> 4) * 8;
#pragma unroll
  for (int h = 0; h < NH; ++h) {
    if (head == h) {
      if (h == 0) {
#pragma unroll
        for (int c = 0; c < 8; ++c)
#pragma unroll
          for (int e = 0; e < 8; ++e) sO[qh][qn][c * 16 + dhi + e] = OT[c][e] * inv;
      } else {
#pragma unroll
        for (int c = 0; c < 8; ++c)
#pragma unroll
          for (int e = 0; e < 8; ++e) sO[qh][qn][c * 16 + dhi + e] += OT[c][e] * inv;
      }
    }
    __syncthreads();
  }

  // Mean over the 4 heads, store 32x128 result
  for (int idx = tid; idx < 32 * TOT; idx += 256) {
    int row = idx >> 7, dim = idx & 127;
    og[(size_t)(blockIdx.x * 32 + row) * TOT + dim] = 0.25f * sO[row >> 4][row & 15][dim];
  }
}

// ---------------- Kernel 3: gates r,z from ax; build x2^T = concat(input, r*hx)^T ----------------
__global__ void mid_kernel(const float* __restrict__ in, const float* __restrict__ hx,
                           const float* __restrict__ ax,
                           const float* __restrict__ Wr, const float* __restrict__ br,
                           const float* __restrict__ Wz, const float* __restrict__ bz,
                           float* __restrict__ zout, __bf16* __restrict__ x2t) {
  int n = blockIdx.x;
  int d = threadIdx.x; // 64
  __shared__ float sax[TOT];
  sax[d]      = ax[(size_t)n * TOT + d];
  sax[d + 64] = ax[(size_t)n * TOT + 64 + d];
  __syncthreads();
  float r = br[d], z = bz[d];
#pragma unroll 4
  for (int t = 0; t < TOT; ++t) {
    float a = sax[t];
    r = fmaf(a, Wr[(size_t)t * HID + d], r);
    z = fmaf(a, Wz[(size_t)t * HID + d], z);
  }
  r = 1.f / (1.f + __expf(-r));
  z = 1.f / (1.f + __expf(-z));
  zout[(size_t)n * HID + d] = z;
  x2t[(size_t)d * NN + n]        = (__bf16)in[(size_t)n * INW + d];
  x2t[(size_t)(64 + d) * NN + n] = (__bf16)(r * hx[(size_t)n * HID + d]);
}

// ---------------- Kernel 4: h = tanh(a2 @ Wh + bh); out = z*hx + (1-z)*h ----------------
__global__ void final_kernel(const float* __restrict__ hx, const float* __restrict__ a2,
                             const float* __restrict__ Wh, const float* __restrict__ bh,
                             const float* __restrict__ zbuf, float* __restrict__ out) {
  int n = blockIdx.x;
  int d = threadIdx.x; // 64
  __shared__ float sa[TOT];
  sa[d]      = a2[(size_t)n * TOT + d];
  sa[d + 64] = a2[(size_t)n * TOT + 64 + d];
  __syncthreads();
  float h = bh[d];
#pragma unroll 4
  for (int t = 0; t < TOT; ++t) h = fmaf(sa[t], Wh[(size_t)t * HID + d], h);
  h = tanhf(h);
  float z = zbuf[(size_t)n * HID + d];
  out[(size_t)n * HID + d] = z * hx[(size_t)n * HID + d] + (1.f - z) * h;
}

extern "C" void kernel_launch(void* const* d_in, const int* in_sizes, int n_in,
                              void* d_out, int out_size, void* d_ws, size_t ws_size,
                              hipStream_t stream) {
  const float* in = (const float*)d_in[0];
  const float* hx = (const float*)d_in[1];
  const float* Wq = (const float*)d_in[2];
  const float* Wk = (const float*)d_in[3];
  const float* Wr = (const float*)d_in[4];
  const float* br = (const float*)d_in[5];
  const float* Wz = (const float*)d_in[6];
  const float* bz = (const float*)d_in[7];
  const float* Wh = (const float*)d_in[8];
  const float* bh = (const float*)d_in[9];
  float* out = (float*)d_out;

  // Workspace layout (bytes)
  char* ws = (char*)d_ws;
  size_t off = 0;
  __bf16* xt   = (__bf16*)(ws + off); off += (size_t)TOT * NN * 2;        // 2 MB  (x^T)
  __bf16* qb   = (__bf16*)(ws + off); off += (size_t)NH * NN * DH * 2;    // 2 MB
  __bf16* kb   = (__bf16*)(ws + off); off += (size_t)NH * NN * DH * 2;    // 2 MB
  float*  ax   = (float*)(ws + off);  off += (size_t)NN * TOT * 4;        // 4 MB
  float*  zbuf = (float*)(ws + off);  off += (size_t)NN * HID * 4;        // 2 MB
  __bf16* x2t  = (__bf16*)(ws + off); off += (size_t)TOT * NN * 2;        // 2 MB  (x2^T)
  float*  a2   = (float*)(ws + off);  off += (size_t)NN * TOT * 4;        // 4 MB

  concat_t_kernel<<<NN, TOT, 0, stream>>>(in, hx, xt);
  qk_kernel<<<NN, TOT, 0, stream>>>(in, hx, Wq, Wk, qb, kb);
  flash_kernel<<<NN / 32, 256, 0, stream>>>(qb, kb, xt, ax);             // ax = adj @ x
  mid_kernel<<<NN, HID, 0, stream>>>(in, hx, ax, Wr, br, Wz, bz, zbuf, x2t);
  flash_kernel<<<NN / 32, 256, 0, stream>>>(qb, kb, x2t, a2);            // a2 = adj @ x2
  final_kernel<<<NN, HID, 0, stream>>>(hx, a2, Wh, bh, zbuf, out);
}
